// MultiHeadAttentionWithBias_19198503813378
// MI455X (gfx1250) — matrix operand order repaired and checked by reading, not compiled, not run on hardware
//
#include <hip/hip_runtime.h>
#include <hip/hip_bf16.h>

// ---------------------------------------------------------------------------
// MultiHeadAttentionWithBias forward for MI455X (gfx1250, wave32, WMMA)
// B=2, S=2048, E=1024, H=16, D=64, SCALE = 0.125
// Pipeline:
//   1. cvt_bf16:      query f32 -> bf16
//   2. transpose_cvt: W[n][k] f32 -> WT[k][n] bf16 (x4)
//   3. qkv_gemm:      bf16 WMMA GEMM, async-to-LDS double-buffered staging,
//                     writes Q/K/V bf16 [b][h][s][d]
//   4. attn:          flash attention; S^T = K*Q^T trick (P -> A-frag is a pure
//                     in-lane pack); async double-buffered K/V tiles
//   5. out_gemm:      bf16 WMMA GEMM + bias -> f32 d_out
// ---------------------------------------------------------------------------

typedef __attribute__((ext_vector_type(16))) __bf16 v16bf;
typedef __attribute__((ext_vector_type(8)))  float  f32x8;
typedef unsigned short u16;

#define BB 2
#define SS 2048
#define EE 1024
#define HH 16
#define DD 64
#define SCALE 0.125f

union ABfrag { v16bf v; uint4 q[2]; };

// gfx1250 async global->LDS copy (ASYNCcnt tracked). lds = 32-bit LDS byte
// address (low 32 bits of generic shared pointer), g = 64-bit global address.
#define ASYNC_LOAD_B128(lds, g) \
  asm volatile("global_load_async_to_lds_b128 %0, %1, off" :: "v"(lds), "v"(g) : "memory")
#define WAIT_ASYNC0() asm volatile("s_wait_asynccnt 0" ::: "memory")

__device__ inline u16 f2bf_u16(float f) {
  union { __bf16 b; u16 u; } c;
  c.b = (__bf16)f;                 // native gfx1250 f32->bf16 convert
  return c.u;
}
__device__ inline __bf16 f2bf(float f) { return (__bf16)f; }
__device__ inline unsigned ldsaddr(const void* p) { return (unsigned)(size_t)p; }

// ---------------------------------------------------------------------------
// 1) elementwise f32 -> bf16
// ---------------------------------------------------------------------------
__global__ __launch_bounds__(256) void cvt_bf16(const float* __restrict__ x,
                                                u16* __restrict__ y, int n) {
  int i = (blockIdx.x * 256 + threadIdx.x) * 4;
  if (i + 3 < n) {
    float4 v = *(const float4*)(x + i);
    y[i + 0] = f2bf_u16(v.x);
    y[i + 1] = f2bf_u16(v.y);
    y[i + 2] = f2bf_u16(v.z);
    y[i + 3] = f2bf_u16(v.w);
  }
}

// ---------------------------------------------------------------------------
// 2) W[n][k] f32 -> WT[k][n] bf16, 32x32 LDS tile transpose, block (32,8)
// ---------------------------------------------------------------------------
__global__ __launch_bounds__(256) void transpose_cvt(const float* __restrict__ W,
                                                     u16* __restrict__ WT) {
  __shared__ float tile[32][33];
  int tx = threadIdx.x, ty = threadIdx.y;
  int k_in = blockIdx.x * 32 + tx;
  int n_base = blockIdx.y * 32;
#pragma unroll
  for (int j = 0; j < 4; ++j)
    tile[ty + j * 8][tx] = W[(size_t)(n_base + ty + j * 8) * EE + k_in];
  __syncthreads();
  int n_out = blockIdx.y * 32 + tx;
  int k_base = blockIdx.x * 32;
#pragma unroll
  for (int j = 0; j < 4; ++j)
    WT[(size_t)(k_base + ty + j * 8) * EE + n_out] = f2bf_u16(tile[tx][ty + j * 8]);
}

// ---------------------------------------------------------------------------
// 3) fused QKV GEMM: C[m][n] = sum_k Abf[m][k]*WT[k][n] + bias[n]
//    M=4096 K=1024 N=3072; block tile 128x128, 8 waves (2x4), wave 64x32.
//    Async-to-LDS double-buffered staging, one barrier per K-step.
// ---------------------------------------------------------------------------
__global__ __launch_bounds__(256) void qkv_gemm(
    const u16* __restrict__ Abf,
    const u16* __restrict__ WqT, const u16* __restrict__ WkT, const u16* __restrict__ WvT,
    const float* __restrict__ bq, const float* __restrict__ bk, const float* __restrict__ bv,
    u16* __restrict__ Qb, u16* __restrict__ Kb, u16* __restrict__ Vb) {
  __shared__ u16 As[2][128][40];   // 32 K + 8 pad
  __shared__ u16 Bs[2][32][136];   // 128 N + 8 pad

  const int tid = threadIdx.x;
  const int wid = tid >> 5, lane = tid & 31;
  const int hi = lane >> 4, lo = lane & 15;
  const int wrow = wid >> 2, wcol = wid & 3;
  const int m0 = blockIdx.y * 128;
  const int n0 = blockIdx.x * 128;
  const int which = n0 >> 10;
  const u16*   WT   = (which == 0) ? WqT : (which == 1) ? WkT : WvT;
  const float* bias = (which == 0) ? bq  : (which == 1) ? bk  : bv;
  u16*         dst  = (which == 0) ? Qb  : (which == 1) ? Kb  : Vb;
  const int nw = n0 & 1023;

  const int ar = tid >> 1, ah = tid & 1;     // A staging coords
  const int br = tid >> 3, bs2 = tid & 7;    // B staging coords

  auto stage = [&](int buf, int k0) {
    const u16* ga = Abf + (size_t)(m0 + ar) * EE + k0 + ah * 16;
    ASYNC_LOAD_B128(ldsaddr(&As[buf][ar][ah * 16]),     ga);
    ASYNC_LOAD_B128(ldsaddr(&As[buf][ar][ah * 16 + 8]), ga + 8);
    const u16* gb = WT + (size_t)(k0 + br) * EE + nw + bs2 * 16;
    ASYNC_LOAD_B128(ldsaddr(&Bs[buf][br][bs2 * 16]),     gb);
    ASYNC_LOAD_B128(ldsaddr(&Bs[buf][br][bs2 * 16 + 8]), gb + 8);
  };

  const f32x8 vzero = {0.f, 0.f, 0.f, 0.f, 0.f, 0.f, 0.f, 0.f};
  f32x8 acc[4][2];
#pragma unroll
  for (int i = 0; i < 4; ++i)
#pragma unroll
    for (int j = 0; j < 2; ++j) acc[i][j] = vzero;

  stage(0, 0);
  WAIT_ASYNC0();
  __syncthreads();

  for (int k0 = 0; k0 < EE; k0 += 32) {
    const int buf = (k0 >> 5) & 1;
    if (k0 + 32 < EE) stage(buf ^ 1, k0 + 32);   // overlap with WMMA below

    ABfrag af[4], bf[2];
#pragma unroll
    for (int i = 0; i < 4; ++i) {
      int row = wrow * 64 + i * 16 + lo;
      int col = hi * 8;
      af[i].q[0] = *(const uint4*)&As[buf][row][col];
      af[i].q[1] = *(const uint4*)&As[buf][row][col + 16];
    }
#pragma unroll
    for (int j = 0; j < 2; ++j) {
      int col = wcol * 32 + j * 16;
#pragma unroll
      for (int s = 0; s < 16; ++s)
        bf[j].v[s] = *(const __bf16*)&Bs[buf][(s & 7) + hi * 8 + (s >> 3) * 16][col + lo];
    }
#pragma unroll
    for (int i = 0; i < 4; ++i)
#pragma unroll
      for (int j = 0; j < 2; ++j)
        acc[i][j] = __builtin_amdgcn_wmma_f32_16x16x32_bf16(
            false, af[i].v, false, bf[j].v, (short)0, acc[i][j], false, false);

    WAIT_ASYNC0();
    __syncthreads();
  }

#pragma unroll
  for (int i = 0; i < 4; ++i)
#pragma unroll
    for (int j = 0; j < 2; ++j)
#pragma unroll
      for (int r = 0; r < 8; ++r) {
        int m = m0 + wrow * 64 + i * 16 + r + hi * 8;
        int n = nw + wcol * 32 + j * 16 + lo;
        float v = acc[i][j][r] + bias[n];
        int bb = m >> 11, s = m & 2047;
        int hh = n >> 6,  d = n & 63;
        dst[(((size_t)bb * HH + hh) * SS + s) * DD + d] = f2bf_u16(v);
      }
}

// ---------------------------------------------------------------------------
// 4) flash attention. Grid (S/128, H, B). 8 waves; wave owns 16 q rows.
//    Scores computed transposed (M=kpos, N=q): softmax stats per lane, P packs
//    in-lane into the A-fragment. K/V tiles async double-buffered; bias+mask
//    fused into a double-buffered f32 LDS tile. One barrier per key tile.
// ---------------------------------------------------------------------------
#define QT_OFF 0
#define KS_OFF 17408                         // 64*136*2
#define VS_OFF (17408 + 18432)               // + 2*64*72*2
#define BT_OFF (17408 + 18432 + 18432)
#define SMEM_BYTES (17408 + 18432 + 18432 + 69632)   // + 2*128*68*4 = 123904

__global__ __launch_bounds__(256) void attn(
    const u16* __restrict__ Qb, const u16* __restrict__ Kb, const u16* __restrict__ Vb,
    const float* __restrict__ biasg, const int* __restrict__ maskg,
    u16* __restrict__ Ob) {
  extern __shared__ __align__(16) char smem[];
  u16 (*Qt)[136]     = (u16 (*)[136])    (smem + QT_OFF);  // Q^T [d=64][q=128]
  u16 (*Ks)[64][72]  = (u16 (*)[64][72]) (smem + KS_OFF);  // [buf][sk][d]
  u16 (*Vs)[64][72]  = (u16 (*)[64][72]) (smem + VS_OFF);  // [buf][sk][d]
  float (*Bt)[128][68] = (float (*)[128][68])(smem + BT_OFF); // [buf][q][k]

  const int b = blockIdx.z, h = blockIdx.y;
  const int q0 = blockIdx.x * 128;
  const size_t base = ((size_t)b * HH + h) * SS * DD;
  const u16* Qp = Qb + base;
  const u16* Kp = Kb + base;
  const u16* Vp = Vb + base;
  const float* biasp = biasg + (size_t)b * SS * SS;
  const int*   maskp = maskg + (size_t)b * SS * SS;

  const int tid = threadIdx.x;
  const int wid = tid >> 5, lane = tid & 31;
  const int hi = lane >> 4, lo = lane & 15;

  auto stage_kv = [&](int buf, int kt) {
    int r = tid >> 2, seg = tid & 3;
    const u16* gk = Kp + (size_t)(kt + r) * DD + seg * 16;
    ASYNC_LOAD_B128(ldsaddr(&Ks[buf][r][seg * 16]),     gk);
    ASYNC_LOAD_B128(ldsaddr(&Ks[buf][r][seg * 16 + 8]), gk + 8);
    const u16* gv = Vp + (size_t)(kt + r) * DD + seg * 16;
    ASYNC_LOAD_B128(ldsaddr(&Vs[buf][r][seg * 16]),     gv);
    ASYNC_LOAD_B128(ldsaddr(&Vs[buf][r][seg * 16 + 8]), gv + 8);
  };
  auto stage_bias = [&](int buf, int kt) {
    int q = tid >> 1, seg = tid & 1;
    const float* bsrc = biasp + (size_t)(q0 + q) * SS + kt + seg * 32;
    const int*   msrc = maskp + (size_t)(q0 + q) * SS + kt + seg * 32;
#pragma unroll
    for (int j = 0; j < 8; ++j) {
      float4 bv4 = ((const float4*)bsrc)[j];
      int4   mv4 = ((const int4*)msrc)[j];
      float4 o;
      o.x = (mv4.x > 0) ? bv4.x : -__builtin_inff();
      o.y = (mv4.y > 0) ? bv4.y : -__builtin_inff();
      o.z = (mv4.z > 0) ? bv4.z : -__builtin_inff();
      o.w = (mv4.w > 0) ? bv4.w : -__builtin_inff();
      *(float4*)&Bt[buf][q][seg * 32 + j * 4] = o;
    }
  };

  // ---- stage Q^T once ----
  {
    int q = tid >> 1, h2 = tid & 1;
    const u16* src = Qp + (size_t)(q0 + q) * DD + h2 * 32;
    union { uint4 v4[4]; u16 u[32]; } tmp;
#pragma unroll
    for (int i = 0; i < 4; ++i) tmp.v4[i] = ((const uint4*)src)[i];
#pragma unroll
    for (int d = 0; d < 32; ++d) Qt[h2 * 32 + d][q] = tmp.u[d];
  }
  stage_kv(0, 0);
  stage_bias(0, 0);
  WAIT_ASYNC0();
  __syncthreads();

  // Q as B-operand fragments (lane%16 = q column, elements = d, 16 q columns of this wave)
  ABfrag qf[2];
#pragma unroll
  for (int ds2 = 0; ds2 < 2; ++ds2) {
    int drow = ds2 * 32 + lane;
    int col = wid * 16;
#pragma unroll
    for (int s = 0; s < 16; ++s)
      qf[ds2].v[s] = *(const __bf16*)&Qt[ds2 * 32 + (s & 7) + hi * 8 + (s >> 3) * 16][col + lo];
  }

  const f32x8 vzero = {0.f, 0.f, 0.f, 0.f, 0.f, 0.f, 0.f, 0.f};
  f32x8 accO[4];
#pragma unroll
  for (int t = 0; t < 4; ++t) accO[t] = vzero;
  float m_run = -__builtin_inff();
  float l_run = 0.f;
  const int qlane = wid * 16 + lo;

  for (int kt = 0; kt < SS; kt += 64) {
    const int buf = (kt >> 6) & 1;
    if (kt + 64 < SS) {          // prefetch next tile into the other buffer
      stage_kv(buf ^ 1, kt + 64);
      stage_bias(buf ^ 1, kt + 64);
    }

    // ---- S^T = K * Q^T : M=kpos(4 tiles), N=q(16), K=d(2x32) ----
    f32x8 sc[4];
#pragma unroll
    for (int t = 0; t < 4; ++t) sc[t] = vzero;
#pragma unroll
    for (int ds2 = 0; ds2 < 2; ++ds2) {
#pragma unroll
      for (int t = 0; t < 4; ++t) {
        ABfrag kf;
        int row = t * 16 + lo;
        int col = ds2 * 32 + hi * 8;
        kf.q[0] = *(const uint4*)&Ks[buf][row][col];
        kf.q[1] = *(const uint4*)&Ks[buf][row][col + 16];
        sc[t] = __builtin_amdgcn_wmma_f32_16x16x32_bf16(
            false, kf.v, false, qf[ds2].v, (short)0, sc[t], false, false);
      }
    }

    // ---- scale + bias + online softmax (per-lane stats, q = qlane) ----
    float tmax = -__builtin_inff();
#pragma unroll
    for (int t = 0; t < 4; ++t)
#pragma unroll
      for (int r = 0; r < 8; ++r) {
        float s = sc[t][r] * SCALE + Bt[buf][qlane][t * 16 + r + hi * 8];
        sc[t][r] = s;
        tmax = fmaxf(tmax, s);
      }
    tmax = fmaxf(tmax, __shfl_xor(tmax, 16));
    float mnew = fmaxf(m_run, tmax);
    float alpha = (mnew == -__builtin_inff()) ? 1.f : __expf(m_run - mnew);
    float lpart = 0.f;
#pragma unroll
    for (int t = 0; t < 4; ++t)
#pragma unroll
      for (int r = 0; r < 8; ++r) {
        float s = sc[t][r];
        float pv = (s == -__builtin_inff()) ? 0.f : __expf(s - mnew);
        sc[t][r] = pv;
        lpart += pv;
      }
    lpart += __shfl_xor(lpart, 16);
    l_run = l_run * alpha + lpart;
    m_run = mnew;

#pragma unroll
    for (int r = 0; r < 8; ++r) {
      float aq = __shfl(alpha, r + hi * 8);
#pragma unroll
      for (int nt = 0; nt < 4; ++nt) accO[nt][r] *= aq;
    }

    // ---- O += P * V (P A-frag = in-lane pack of sc tiles) ----
#pragma unroll
    for (int kk = 0; kk < 2; ++kk) {
      ABfrag pf;
      int t0 = kk * 2, t1 = kk * 2 + 1;
#pragma unroll
      for (int i = 0; i < 8; ++i) {
        pf.v[i]     = f2bf(sc[t0][i]);
        pf.v[i + 8] = f2bf(sc[t1][i]);
      }
#pragma unroll
      for (int nt = 0; nt < 4; ++nt) {
        ABfrag vf;
#pragma unroll
        for (int s = 0; s < 16; ++s)
          vf.v[s] = *(const __bf16*)&Vs[buf][kk * 32 + (s & 7) + hi * 8 + (s >> 3) * 16][nt * 16 + lo];
        accO[nt] = __builtin_amdgcn_wmma_f32_16x16x32_bf16(
            false, pf.v, false, vf.v, (short)0, accO[nt], false, false);
      }
    }

    WAIT_ASYNC0();     // my next-tile async copies done
    __syncthreads();   // everyone done computing buf + staging buf^1
  }

  // ---- normalize + store to [b][s][h*64+d] bf16 ----
#pragma unroll
  for (int r = 0; r < 8; ++r) {
    float lq = __shfl(l_run, r + hi * 8);
    float inv = (lq > 0.f) ? (1.f / lq) : 0.f;
    int q = q0 + wid * 16 + r + hi * 8;
#pragma unroll
    for (int nt = 0; nt < 4; ++nt) {
      int d = nt * 16 + lo;
      float o = accO[nt][r] * inv;
      Ob[((size_t)b * SS + q) * EE + h * DD + d] = f2bf_u16(o);
    }
  }
}

// ---------------------------------------------------------------------------
// 5) output projection: out[m][n] = sum_k Ao[m][k]*WoT[k][n] + bo[n], f32 out
// ---------------------------------------------------------------------------
__global__ __launch_bounds__(256) void out_gemm(
    const u16* __restrict__ Abf, const u16* __restrict__ WT,
    const float* __restrict__ bo, float* __restrict__ Cout) {
  __shared__ u16 As[2][128][40];
  __shared__ u16 Bs[2][32][136];

  const int tid = threadIdx.x;
  const int wid = tid >> 5, lane = tid & 31;
  const int hi = lane >> 4, lo = lane & 15;
  const int wrow = wid >> 2, wcol = wid & 3;
  const int m0 = blockIdx.y * 128;
  const int n0 = blockIdx.x * 128;

  const int ar = tid >> 1, ah = tid & 1;
  const int br = tid >> 3, bs2 = tid & 7;

  auto stage = [&](int buf, int k0) {
    const u16* ga = Abf + (size_t)(m0 + ar) * EE + k0 + ah * 16;
    ASYNC_LOAD_B128(ldsaddr(&As[buf][ar][ah * 16]),     ga);
    ASYNC_LOAD_B128(ldsaddr(&As[buf][ar][ah * 16 + 8]), ga + 8);
    const u16* gb = WT + (size_t)(k0 + br) * EE + n0 + bs2 * 16;
    ASYNC_LOAD_B128(ldsaddr(&Bs[buf][br][bs2 * 16]),     gb);
    ASYNC_LOAD_B128(ldsaddr(&Bs[buf][br][bs2 * 16 + 8]), gb + 8);
  };

  const f32x8 vzero = {0.f, 0.f, 0.f, 0.f, 0.f, 0.f, 0.f, 0.f};
  f32x8 acc[4][2];
#pragma unroll
  for (int i = 0; i < 4; ++i)
#pragma unroll
    for (int j = 0; j < 2; ++j) acc[i][j] = vzero;

  stage(0, 0);
  WAIT_ASYNC0();
  __syncthreads();

  for (int k0 = 0; k0 < EE; k0 += 32) {
    const int buf = (k0 >> 5) & 1;
    if (k0 + 32 < EE) stage(buf ^ 1, k0 + 32);

    ABfrag af[4], bf[2];
#pragma unroll
    for (int i = 0; i < 4; ++i) {
      int row = wrow * 64 + i * 16 + lo;
      int col = hi * 8;
      af[i].q[0] = *(const uint4*)&As[buf][row][col];
      af[i].q[1] = *(const uint4*)&As[buf][row][col + 16];
    }
#pragma unroll
    for (int j = 0; j < 2; ++j) {
      int col = wcol * 32 + j * 16;
#pragma unroll
      for (int s = 0; s < 16; ++s)
        bf[j].v[s] = *(const __bf16*)&Bs[buf][(s & 7) + hi * 8 + (s >> 3) * 16][col + lo];
    }
#pragma unroll
    for (int i = 0; i < 4; ++i)
#pragma unroll
      for (int j = 0; j < 2; ++j)
        acc[i][j] = __builtin_amdgcn_wmma_f32_16x16x32_bf16(
            false, af[i].v, false, bf[j].v, (short)0, acc[i][j], false, false);

    WAIT_ASYNC0();
    __syncthreads();
  }

#pragma unroll
  for (int i = 0; i < 4; ++i)
#pragma unroll
    for (int j = 0; j < 2; ++j)
#pragma unroll
      for (int r = 0; r < 8; ++r) {
        int m = m0 + wrow * 64 + i * 16 + r + hi * 8;
        int n = n0 + wcol * 32 + j * 16 + lo;
        Cout[(size_t)m * EE + n] = acc[i][j][r] + bo[n];
      }
}

// ---------------------------------------------------------------------------
extern "C" void kernel_launch(void* const* d_in, const int* in_sizes, int n_in,
                              void* d_out, int out_size, void* d_ws, size_t ws_size,
                              hipStream_t stream) {
  const float* query = (const float*)d_in[0];
  const float* abias = (const float*)d_in[1];
  const int*   amask = (const int*)d_in[2];
  const float* Wq = (const float*)d_in[3];
  const float* bq = (const float*)d_in[4];
  const float* Wk = (const float*)d_in[5];
  const float* bk = (const float*)d_in[6];
  const float* Wv = (const float*)d_in[7];
  const float* bv = (const float*)d_in[8];
  const float* Wo = (const float*)d_in[9];
  const float* bo = (const float*)d_in[10];
  float* out = (float*)d_out;

  char* p = (char*)d_ws;
  auto carve = [&](size_t bytes) -> char* {
    char* r = p;
    p += (bytes + 255) & ~(size_t)255;
    return r;
  };
  const size_t ACT = (size_t)BB * SS * EE * 2;   // 8 MB bf16 activations
  const size_t WTB = (size_t)EE * EE * 2;        // 2 MB bf16 weight
  u16* qbf = (u16*)carve(ACT);
  u16* WqT = (u16*)carve(WTB);
  u16* WkT = (u16*)carve(WTB);
  u16* WvT = (u16*)carve(WTB);
  u16* WoT = (u16*)carve(WTB);
  u16* Qb  = (u16*)carve(ACT);
  u16* Kb  = (u16*)carve(ACT);
  u16* Vb  = (u16*)carve(ACT);
  u16* Ao  = (u16*)carve(ACT);

  cvt_bf16<<<dim3((BB * SS * EE) / 1024), 256, 0, stream>>>(query, qbf, BB * SS * EE);

  dim3 tb(32, 8), tg(EE / 32, EE / 32);
  transpose_cvt<<<tg, tb, 0, stream>>>(Wq, WqT);
  transpose_cvt<<<tg, tb, 0, stream>>>(Wk, WkT);
  transpose_cvt<<<tg, tb, 0, stream>>>(Wv, WvT);
  transpose_cvt<<<tg, tb, 0, stream>>>(Wo, WoT);

  qkv_gemm<<<dim3(3 * EE / 128, BB * SS / 128), 256, 0, stream>>>(
      qbf, WqT, WkT, WvT, bq, bk, bv, Qb, Kb, Vb);

  attn<<<dim3(SS / 128, HH, BB), 256, SMEM_BYTES, stream>>>(
      Qb, Kb, Vb, abias, amask, Ao);

  out_gemm<<<dim3(EE / 128, BB * SS / 128), 256, 0, stream>>>(Ao, WoT, bo, out);
}
